// SSD_60593398612320
// MI455X (gfx1250) — matrix operand order repaired and checked by reading, not compiled, not run on hardware
//
#include <hip/hip_runtime.h>

typedef __bf16 bf16_t;
typedef __attribute__((ext_vector_type(16))) __bf16 v16bf;
typedef __attribute__((ext_vector_type(8)))  float  v8f;

#define BATCH   8
#define ATOT    21760
#define FG      20
#define NCLS    21
#define THRESH  0.01f
#define NMS_TH  0.45f
#define TOPK    400
#define POSTN   100
#define CAP     4096
#define NPIX    128         // pixels per block (8 x 16-pixel WMMA sub-tiles)
#define NSUB    (NPIX / 16)

#define AS1 __attribute__((address_space(1)))
#define AS3 __attribute__((address_space(3)))

// ---------------------------------------------------------------------------
// 1) Repack OIHW f32 weights (cls 0..83, box 84..99, pad to 128) into per-lane
//    bf16 WMMA A-fragments: [m_tile(8)][tap(9)][kc][lane(32)][j(16)]
//    lane L holds M = L&15; element j holds K = (j&7) + ((j>>3)<<4) + ((L>>4)<<3)
// ---------------------------------------------------------------------------
__global__ void prep_weights(const float* __restrict__ cls_w,
                             const float* __restrict__ box_w,
                             bf16_t* __restrict__ dst, int Cin, int ntot)
{
    int e = blockIdx.x * blockDim.x + threadIdx.x;
    if (e >= ntot) return;
    int KC   = Cin >> 5;
    int j    = e & 15;
    int lane = (e >> 4) & 31;
    int kc   = (e >> 9) % KC;
    int t    = (e >> 9) / KC;
    int tap  = t % 9;
    int mt   = t / 9;
    int m    = lane & 15;
    int k    = (j & 7) + ((j >> 3) << 4) + ((lane >> 4) << 3);
    int cin  = (kc << 5) + k;
    int c    = mt * 16 + m;
    int ky   = tap / 3, kx = tap % 3;
    float v = 0.f;
    if (c < 84)       v = cls_w[(((long)c * Cin + cin) * 3 + ky) * 3 + kx];
    else if (c < 100) v = box_w[(((long)(c - 84) * Cin + cin) * 3 + ky) * 3 + kx];
    dst[e] = (bf16_t)v;
}

// ---------------------------------------------------------------------------
// 2) Implicit-GEMM 3x3 conv, double-buffered async pipeline.
//    Block = 256 thr = 8 waves, owns 128 pixels. Chunk = (tap, 32-cin slice).
//    Steady state per chunk c:
//      s_wait_asynccnt 0 ; barrier          (chunk c staged by all waves)
//      convert Fstage[c&1] -> Bfrag[c&1]    (f32 -> bf16, fragment order)
//      barrier ; issue async for chunk c+1  (into buffers (c+1)&1)
//      waves 0..6: 1 A-load + 8 x (32B LDS B-load + WMMA)   <- overlaps c+1 loads
// ---------------------------------------------------------------------------
__global__ __launch_bounds__(256)
void conv_wmma(const float* __restrict__ feat, const bf16_t* __restrict__ wfrag,
               const float* __restrict__ clsb, const float* __restrict__ boxb,
               float* __restrict__ cls_out, float* __restrict__ box_raw,
               int Cin, int H, int W, int abase)
{
    const int HW   = H * W;
    const int KC   = Cin >> 5;
    const int NCH  = 9 * KC;                 // total K chunks (tap-major)
    const int lane = threadIdx.x & 31;
    const int wave = threadIdx.x >> 5;
    const int p0   = blockIdx.x * NPIX;

    __shared__ float Fstage[2][32 * NPIX];             // 2 x 16 KB async targets
    __shared__ alignas(32) bf16_t Bfrag[2][NSUB][512]; // 2 x 8 KB fragment tiles

    v8f acc[NSUB] = {};

    // issue all async loads for chunk c into buffer `buf`
    auto issue = [&](int c, int buf) {
        int tap = c / KC, kc = c - tap * KC;
        int ky = tap / 3 - 1, kx = tap % 3 - 1;
        #pragma unroll
        for (int r = 0; r < (32 * NPIX) / 256; ++r) {
            int e = r * 256 + threadIdx.x;             // e = k*NPIX + n
            int k = e >> 7, n = e & (NPIX - 1);
            int p = p0 + n;
            int b = p / HW, rem = p - b * HW;
            int y = rem / W, x = rem - y * W;
            int yy = y + ky, xx = x + kx;
            if (yy >= 0 && yy < H && xx >= 0 && xx < W) {
                int cin = (kc << 5) + k;
                const float* g = feat + (((long)b * Cin + cin) * H + yy) * W + xx;
                __builtin_amdgcn_global_load_async_to_lds_b32(
                    (AS1 int*)g, (AS3 int*)&Fstage[buf][e], 0, 0);
            } else {
                Fstage[buf][e] = 0.f;                  // SAME padding
            }
        }
    };

    issue(0, 0);                                       // prologue

    for (int c = 0; c < NCH; ++c) {
        const int buf = c & 1;
        // each wave drains its own ASYNCcnt, then barrier makes all waves'
        // chunk-c data visible to everyone
        asm volatile("s_wait_asynccnt 0x0" ::: "memory");
        __syncthreads();

        // convert: thread t owns fragment lane L = t&31 of sub-tile s = t>>5
        {
            int L = threadIdx.x & 31, s = threadIdx.x >> 5;
            #pragma unroll
            for (int j = 0; j < 16; ++j) {
                int k = (j & 7) + ((j >> 3) << 4) + ((L >> 4) << 3); // frag elem j -> K index
                int n = (s << 4) + (L & 15);           // pixel within block
                Bfrag[buf][s][(L << 4) + j] = (bf16_t)Fstage[buf][(k << 7) + n];
            }
        }
        __syncthreads();

        if (c + 1 < NCH) issue(c + 1, buf ^ 1);        // prefetch next chunk

        // compute (wave-uniform branch keeps EXEC all-ones for WMMA)
        int tap = c / KC, kc = c - tap * KC;
        if (wave < 7) {
            const v16bf* ap = (const v16bf*)(wfrag +
                ((((long)wave * 9 + tap) * KC + kc) * 32 + lane) * 16);
            v16bf a = *ap;
            #pragma unroll
            for (int s = 0; s < NSUB; ++s) {
                v16bf bv = *(const v16bf*)&Bfrag[buf][s][lane << 4];
                acc[s] = __builtin_amdgcn_wmma_f32_16x16x32_bf16(
                            false, a, false, bv, (short)0, acc[s], false, false);
            }
        }
    }

    // --- C scatter: VGPR v, lane L -> M = v + 8*(L>>4), N = L&15 ---
    if (wave < 7) {
        #pragma unroll
        for (int s = 0; s < NSUB; ++s) {
            const int n    = (s << 4) + (lane & 15);
            const int p    = p0 + n;
            const int b    = p / HW;
            const int cell = p - b * HW;               // y*W + x
            #pragma unroll
            for (int v = 0; v < 8; ++v) {
                int m = v + ((lane >> 4) << 3);
                int c = wave * 16 + m;
                float val = acc[s][v];
                if (c < 84) {
                    int d = c / 21, cl = c - d * 21;
                    int a = abase + cell * 4 + d;
                    cls_out[((long)b * ATOT + a) * NCLS + cl] = val + clsb[c];
                } else if (c < 100) {
                    int bc = c - 84;
                    int d = bc >> 2, co = bc & 3;
                    int a = abase + cell * 4 + d;
                    box_raw[((long)b * ATOT + a) * 4 + co] = val + boxb[bc];
                }
            }
        }
    }
}

// ---------------------------------------------------------------------------
// 3) softmax over 21 classes + SSD anchor decode
// ---------------------------------------------------------------------------
__global__ void decode_softmax(const float* __restrict__ cls_out,
                               const float* __restrict__ box_raw,
                               float* __restrict__ probs,
                               float* __restrict__ boxes)
{
    int idx = blockIdx.x * blockDim.x + threadIdx.x;
    if (idx >= BATCH * ATOT) return;
    int b = idx / ATOT, a = idx - b * ATOT;

    int base, Hh, step; float s0, s1;
    if      (a < 16384) { base = 0;     Hh = 64; step = 8;  s0 = 51.2f;  s1 = 102.4f; }
    else if (a < 20480) { base = 16384; Hh = 32; step = 16; s0 = 102.4f; s1 = 189.4f; }
    else if (a < 21504) { base = 20480; Hh = 16; step = 32; s0 = 189.4f; s1 = 276.4f; }
    else                { base = 21504; Hh = 8;  step = 64; s0 = 276.4f; s1 = 363.5f; }
    int al = a - base;
    int cell = al >> 2, d = al & 3;
    int y = cell / Hh, x = cell - y * Hh;
    float ax = (x + 0.5f) * step, ay = (y + 0.5f) * step;
    float aw, ah;
    if      (d == 0) { aw = s0; ah = s0; }
    else if (d == 1) { float g = sqrtf(s0 * s1); aw = g; ah = g; }
    else if (d == 2) { float r = sqrtf(2.0f); aw = s0 * r; ah = s0 / r; }
    else             { float r = sqrtf(0.5f); aw = s0 * r; ah = s0 / r; }

    const float* lg = cls_out + ((long)b * ATOT + a) * NCLS;
    float mx = lg[0];
    for (int c = 1; c < NCLS; ++c) mx = fmaxf(mx, lg[c]);
    float ex[NCLS], sum = 0.f;
    for (int c = 0; c < NCLS; ++c) { ex[c] = __expf(lg[c] - mx); sum += ex[c]; }
    float inv = 1.f / sum;
    float* pr = probs + ((long)b * ATOT + a) * FG;
    for (int c = 0; c < FG; ++c) {
        float p = ex[c + 1] * inv;
        pr[c] = (p > THRESH) ? p : 0.f;
    }

    const float* t = box_raw + ((long)b * ATOT + a) * 4;
    float ox = t[0] * 0.1f * aw + ax;
    float oy = t[1] * 0.1f * ah + ay;
    float hw = __expf(t[2] * 0.2f) * aw * 0.5f;
    float hh = __expf(t[3] * 0.2f) * ah * 0.5f;
    float* bo = boxes + ((long)b * ATOT + a) * 4;
    bo[0] = ox - hw; bo[1] = oy - hh; bo[2] = ox + hw; bo[3] = oy + hh;
}

// ---------------------------------------------------------------------------
// 4) Per-image top-400: threshold-append + LDS bitonic sort (descending)
// ---------------------------------------------------------------------------
__global__ __launch_bounds__(1024)
void topk_select(const float* __restrict__ probs,
                 const float* __restrict__ boxes,
                 float* __restrict__ cand)
{
    __shared__ float    ss[CAP];
    __shared__ unsigned sc[CAP];
    __shared__ int      cnt;
    int b = blockIdx.x, t = threadIdx.x;
    if (t == 0) cnt = 0;
    __syncthreads();

    const float* pr = probs + (long)b * ATOT * FG;
    const int total = ATOT * FG;
    for (int i = t; i < total; i += 1024) {
        float s = pr[i];
        if (s > 0.f) {
            int pos = atomicAdd(&cnt, 1);
            if (pos < CAP) { ss[pos] = s; sc[pos] = (unsigned)i; }
        }
    }
    __syncthreads();
    int n = (cnt < CAP) ? cnt : CAP;
    for (int i = t; i < CAP; i += 1024)
        if (i >= n) { ss[i] = -1.f; sc[i] = 0xFFFFFFFFu; }
    __syncthreads();

    for (int k = 2; k <= CAP; k <<= 1) {
        for (int j = k >> 1; j > 0; j >>= 1) {
            for (int i = t; i < CAP; i += 1024) {
                int ixj = i ^ j;
                if (ixj > i) {
                    bool dir = ((i & k) == 0);
                    float si = ss[i], sj = ss[ixj];
                    bool sw = dir ? (si < sj) : (si > sj);
                    if (sw) {
                        ss[i] = sj; ss[ixj] = si;
                        unsigned ci = sc[i]; sc[i] = sc[ixj]; sc[ixj] = ci;
                    }
                }
            }
            __syncthreads();
        }
    }

    for (int s = t; s < TOPK; s += 1024) {
        float* o = cand + ((long)b * TOPK + s) * 6;
        float sv = ss[s];
        if (sv > 0.f) {
            unsigned code = sc[s];
            int a = (int)(code / FG), cl = (int)(code - (unsigned)a * FG);
            const float* bo = boxes + ((long)b * ATOT + a) * 4;
            o[0] = (float)cl; o[1] = sv;
            o[2] = bo[0]; o[3] = bo[1]; o[4] = bo[2]; o[5] = bo[3];
        } else {
            o[0] = -1.f; o[1] = 0.f; o[2] = o[3] = o[4] = o[5] = 0.f;
        }
    }
}

// ---------------------------------------------------------------------------
// 5) Per-image class-wise NMS over 400 sorted candidates, emit 100
// ---------------------------------------------------------------------------
__global__ __launch_bounds__(512)
void nms_kernel(const float* __restrict__ cand,
                float* __restrict__ out_ids,
                float* __restrict__ out_sc,
                float* __restrict__ out_bb)
{
    __shared__ float id_[TOPK], sc_[TOPK], x1_[TOPK], y1_[TOPK], x2_[TOPK], y2_[TOPK], ar_[TOPK];
    __shared__ int   keep_[TOPK], order_[TOPK], nkeep;
    int b = blockIdx.x, t = threadIdx.x;

    for (int i = t; i < TOPK; i += 512) {
        const float* c = cand + ((long)b * TOPK + i) * 6;
        id_[i] = c[0]; sc_[i] = c[1];
        x1_[i] = c[2]; y1_[i] = c[3]; x2_[i] = c[4]; y2_[i] = c[5];
        ar_[i] = fmaxf(x2_[i] - x1_[i], 0.f) * fmaxf(y2_[i] - y1_[i], 0.f);
        keep_[i] = (id_[i] >= 0.f) && (sc_[i] > 0.f);
    }
    __syncthreads();

    for (int i = 0; i < TOPK; ++i) {
        if (keep_[i]) {
            if (t < TOPK && t > i && keep_[t] && id_[t] == id_[i]) {
                float iw = fminf(x2_[i], x2_[t]) - fmaxf(x1_[i], x1_[t]);
                float ih = fminf(y2_[i], y2_[t]) - fmaxf(y1_[i], y1_[t]);
                iw = fmaxf(iw, 0.f); ih = fmaxf(ih, 0.f);
                float inter = iw * ih;
                float iou = inter / (ar_[i] + ar_[t] - inter + 1e-12f);
                if (iou > NMS_TH) keep_[t] = 0;
            }
        }
        __syncthreads();
    }

    if (t == 0) {
        int p = 0;
        for (int i = 0; i < TOPK; ++i) if (keep_[i]) order_[p++] = i;
        nkeep = p;
    }
    __syncthreads();

    for (int s = t; s < POSTN; s += 512) {
        long o = (long)b * POSTN + s;
        if (s < nkeep) {
            int i = order_[s];
            out_ids[o] = id_[i]; out_sc[o] = sc_[i];
            out_bb[o * 4 + 0] = x1_[i]; out_bb[o * 4 + 1] = y1_[i];
            out_bb[o * 4 + 2] = x2_[i]; out_bb[o * 4 + 3] = y2_[i];
        } else {
            out_ids[o] = -1.f; out_sc[o] = -1.f;
            out_bb[o * 4 + 0] = out_bb[o * 4 + 1] = -1.f;
            out_bb[o * 4 + 2] = out_bb[o * 4 + 3] = -1.f;
        }
    }
}

// ---------------------------------------------------------------------------
extern "C" void kernel_launch(void* const* d_in, const int* in_sizes, int n_in,
                              void* d_out, int out_size, void* d_ws, size_t ws_size,
                              hipStream_t stream)
{
    const float* feat[4];
    const float *cls_w[4], *cls_b[4], *box_w[4], *box_b[4];
    for (int i = 0; i < 4; ++i) feat[i] = (const float*)d_in[i];
    for (int i = 0; i < 4; ++i) {
        cls_w[i] = (const float*)d_in[4 + 4 * i];
        cls_b[i] = (const float*)d_in[5 + 4 * i];
        box_w[i] = (const float*)d_in[6 + 4 * i];
        box_b[i] = (const float*)d_in[7 + 4 * i];
    }

    const int  Cin[4]   = {512, 1024, 512, 256};
    const int  Hs[4]    = {64, 32, 16, 8};
    const int  abase[4] = {0, 16384, 20480, 21504};
    const long wf_off[4] = {0, 589824, 1769472, 2359296};  // bf16 elements
    const long wf_total  = 2654208;

    char* ws = (char*)d_ws;
    bf16_t* wfrag  = (bf16_t*)ws;
    float* cls_out = (float*)(ws + wf_total * 2);            // [8][21760][21]
    float* box_raw = cls_out + (long)BATCH * ATOT * NCLS;    // [8][21760][4]
    float* probs   = box_raw + (long)BATCH * ATOT * 4;       // [8][21760][20]
    float* boxes   = probs   + (long)BATCH * ATOT * FG;      // [8][21760][4]
    float* cand    = boxes   + (long)BATCH * ATOT * 4;       // [8][400][6]

    for (int l = 0; l < 4; ++l) {
        int ntot = 36864 * (Cin[l] >> 5);
        prep_weights<<<ntot / 256, 256, 0, stream>>>(
            cls_w[l], box_w[l], wfrag + wf_off[l], Cin[l], ntot);
    }
    for (int l = 0; l < 4; ++l) {
        int HW = Hs[l] * Hs[l];
        int blocks = BATCH * HW / NPIX;
        conv_wmma<<<blocks, 256, 0, stream>>>(
            feat[l], wfrag + wf_off[l], cls_b[l], box_b[l],
            cls_out, box_raw, Cin[l], Hs[l], Hs[l], abase[l]);
    }
    int tot = BATCH * ATOT;
    decode_softmax<<<(tot + 255) / 256, 256, 0, stream>>>(cls_out, box_raw, probs, boxes);
    topk_select<<<BATCH, 1024, 0, stream>>>(probs, boxes, cand);

    float* out = (float*)d_out;
    nms_kernel<<<BATCH, 512, 0, stream>>>(cand,
                                          out,
                                          out + BATCH * POSTN,
                                          out + 2 * BATCH * POSTN);
}
